// STLSTMGNN_45792941310333
// MI455X (gfx1250) — matrix operand order repaired and checked by reading, not compiled, not run on hardware
//
#include <hip/hip_runtime.h>

// ---------------------------------------------------------------------------
// MI455X (gfx1250) fused ST-GAT + LSTM head.
// All matrix math on v_wmma_f32_16x16x32_bf16 (wave32 WMMA, f32 accumulate).
// ---------------------------------------------------------------------------

typedef __attribute__((ext_vector_type(16))) __bf16 v16bf;
typedef __attribute__((ext_vector_type(8)))  __bf16 bf16x8;
typedef __attribute__((ext_vector_type(8)))  float  v8f;

#define A_NODES 22
#define F_IN    16
#define HEADS   4
#define CDIM    128
#define HC      512   // HEADS*CDIM
#define HID     256
#define G4      1024  // 4*HID
#define BATCH   32
#define TSTEPS  256
#define NFRAMES (BATCH*TSTEPS)

__device__ inline v8f vzero8() {
    v8f z;
#pragma unroll
    for (int i = 0; i < 8; ++i) z[i] = 0.f;
    return z;
}

__device__ inline v8f wmma_bf16(v16bf a, v16bf b, v8f c) {
    // D = A(16x32 bf16) * B(32x16 bf16) + C(16x16 f32)
    return __builtin_amdgcn_wmma_f32_16x16x32_bf16(
        /*neg_a=*/false, a, /*neg_b=*/false, b,
        /*c_mod=*/(short)0, c, /*reuse_a=*/false, /*reuse_b=*/false);
}

// A-fragment from row-major bf16 matrix in LDS.
// 16-bit A 16x32 layout (matches packed B): lane<16 -> row=lane, K={0..15};
// lane>=16 -> row=lane-16, K={16..31}.
__device__ inline v16bf load_a_frag(const __bf16* m, int m0, int k0, int ldm, int lane) {
    int row = m0 + (lane & 15);
    int kh  = (lane & 16) ? 16 : 0;
    const __bf16* p = m + row * ldm + k0 + kh;
    bf16x8 lo = *(const bf16x8*)p;
    bf16x8 hi = *(const bf16x8*)(p + 8);
    v16bf a;
#pragma unroll
    for (int i = 0; i < 8; ++i) { a[i] = lo[i]; a[i + 8] = hi[i]; }
    return a;
}

// B-fragment from pre-packed global buffer: frag[fragIdx][lane][16 halves].
__device__ inline v16bf load_b_frag(const __bf16* fragbuf, int fragIdx, int lane) {
    const __bf16* p = fragbuf + ((size_t)fragIdx * 32 + lane) * 16;
    bf16x8 lo = *(const bf16x8*)p;
    bf16x8 hi = *(const bf16x8*)(p + 8);
    v16bf b;
#pragma unroll
    for (int i = 0; i < 8; ++i) { b[i] = lo[i]; b[i + 8] = hi[i]; }
    return b;
}

__device__ inline float sigf(float x) { return 1.f / (1.f + __expf(-x)); }

// ---------------------------------------------------------------------------
// Weight packing: build bf16 B-fragments.
// B layout used here (contiguous K, matches load_a_frag) (32x16, wave32): lane n=lane&15, K = (lane<16?0:16)+e,
// 16 contiguous K-halves per lane.
// ---------------------------------------------------------------------------

// W1: [16,512] row-major, K padded 16->32 with zeros. Frag index = ntile (32).
__global__ void pack_w1(const float* __restrict__ W1, __bf16* __restrict__ dst) {
    int tid = blockIdx.x * blockDim.x + threadIdx.x;
    if (tid >= 32 * 32 * 16) return;
    int e = tid & 15, lane = (tid >> 4) & 31, nt = tid >> 9;
    int k = ((lane & 16) ? 16 : 0) + e;
    int n = nt * 16 + (lane & 15);
    float v = (k < F_IN) ? W1[k * HC + n] : 0.f;
    dst[tid] = (__bf16)v;
}

// W2: [512,128] row-major. Frag index = kt*8+nt (16x8).
__global__ void pack_w2(const float* __restrict__ W2, __bf16* __restrict__ dst) {
    int tid = blockIdx.x * blockDim.x + threadIdx.x;
    if (tid >= 16 * 8 * 32 * 16) return;
    int e = tid & 15, lane = (tid >> 4) & 31, nt = (tid >> 9) & 7, kt = tid >> 12;
    int k = kt * 32 + ((lane & 16) ? 16 : 0) + e;
    int n = nt * 16 + (lane & 15);
    dst[tid] = (__bf16)W2[k * CDIM + n];
}

// Concatenated LSTM weights: B[k][n] = k<Kx ? Wx[n][k] : Wh[n][k-Kx].
// Wx: [1024,Kx], Wh: [1024,Kh] row-major. Frag index = kt*64+nt.
__global__ void pack_cat(const float* __restrict__ Wx, const float* __restrict__ Wh,
                         int Kx, int Ktot, __bf16* __restrict__ dst) {
    int tid = blockIdx.x * blockDim.x + threadIdx.x;
    int total = (Ktot / 32) * 64 * 32 * 16;
    if (tid >= total) return;
    int e = tid & 15, lane = (tid >> 4) & 31, nt = (tid >> 9) & 63, kt = tid >> 15;
    int k = kt * 32 + ((lane & 16) ? 16 : 0) + e;
    int n = nt * 16 + (lane & 15);
    int Kh = Ktot - Kx;
    float v = (k < Kx) ? Wx[n * Kx + k] : Wh[n * Kh + (k - Kx)];
    dst[tid] = (__bf16)v;
}

// ---------------------------------------------------------------------------
// Fused per-frame GAT1 + GAT2 + mean-pool. One block per frame, 8 waves.
// ---------------------------------------------------------------------------
__global__ __launch_bounds__(256) void gat_frame(
    const float* __restrict__ feat, const int* __restrict__ seqlen,
    const __bf16* __restrict__ W1bf,
    const float* __restrict__ att_src1, const float* __restrict__ att_dst1,
    const float* __restrict__ b1,
    const __bf16* __restrict__ W2bf,
    const float* __restrict__ att_src2, const float* __restrict__ att_dst2,
    const float* __restrict__ b2,
    float* __restrict__ pooled)
{
    __shared__ __bf16 xpad[32 * 32];      // padded A for GEMM1
    __shared__ __bf16 h1bf[32 * 512];     // GAT1 projection (bf16)
    __shared__ __bf16 out1bf[32 * 512];   // relu(GAT1 out) = A for GEMM2
    __shared__ float  h2s[32 * 128];      // GAT2 projection (f32)
    __shared__ float  asrc[24 * 4], adst[24 * 4];
    __shared__ float  alpha1[4 * 22 * 22];
    __shared__ float  asrc2[24], adst2[24], alpha2[22 * 22];

    const int tid = threadIdx.x, lane = tid & 31, wave = tid >> 5;
    const int frame = blockIdx.x;
    const float* xf = feat + (size_t)frame * A_NODES * F_IN;

    for (int i = tid; i < 32 * 32; i += 256) xpad[i] = (__bf16)0.f;
    __syncthreads();
    for (int i = tid; i < A_NODES * F_IN; i += 256) {
        int r = i >> 4, c = i & 15;
        xpad[r * 32 + c] = (__bf16)xf[i];
    }
    __syncthreads();

    // ---- GEMM1: (32x32) x (32x512) -> h1, 64 WMMA tiles, K=32 ----
    for (int tt = wave; tt < 64; tt += 8) {
        int mt = tt & 1, nt = tt >> 1;
        v16bf a = load_a_frag(xpad, mt * 16, 0, 32, lane);
        v16bf b = load_b_frag(W1bf, nt, lane);
        v8f acc = vzero8();
        acc = wmma_bf16(a, b, acc);
        int rbase = mt * 16 + ((lane & 16) ? 8 : 0);
        int col   = nt * 16 + (lane & 15);
#pragma unroll
        for (int v = 0; v < 8; ++v) h1bf[(rbase + v) * HC + col] = (__bf16)acc[v];
    }
    __syncthreads();

    // ---- attention 1 (4 heads) ----
    if (tid < A_NODES * HEADS) {
        int j = tid >> 2, hd = tid & 3;
        float s = 0.f, d = 0.f;
        const float* ps = att_src1 + hd * CDIM;
        const float* pd = att_dst1 + hd * CDIM;
        for (int c = 0; c < CDIM; ++c) {
            float hv = (float)h1bf[j * HC + hd * CDIM + c];
            s += hv * ps[c]; d += hv * pd[c];
        }
        asrc[j * 4 + hd] = s; adst[j * 4 + hd] = d;
    }
    __syncthreads();
    if (tid < A_NODES * HEADS) {
        int i = tid >> 2, hd = tid & 3;
        float di = adst[i * 4 + hd];
        float ev[A_NODES], mx = -1e30f;
        for (int j = 0; j < A_NODES; ++j) {
            float z = di + asrc[j * 4 + hd];
            z = (z > 0.f) ? z : 0.2f * z;            // leaky_relu 0.2
            ev[j] = z; mx = fmaxf(mx, z);
        }
        float sum = 0.f;
        for (int j = 0; j < A_NODES; ++j) { float e = __expf(ev[j] - mx); ev[j] = e; sum += e; }
        float inv = 1.f / sum;
        for (int j = 0; j < A_NODES; ++j) alpha1[(hd * 22 + i) * 22 + j] = ev[j] * inv;
    }
    __syncthreads();

    // ---- aggregate + bias + relu -> out1 ----
    for (int idx = tid; idx < A_NODES * HC; idx += 256) {
        int i = idx >> 9, hc = idx & 511, hd = hc >> 7;
        const float* al = &alpha1[(hd * 22 + i) * 22];
        float acc = 0.f;
        for (int j = 0; j < A_NODES; ++j) acc += al[j] * (float)h1bf[j * HC + hc];
        acc += b1[hc];
        out1bf[i * HC + hc] = (__bf16)fmaxf(acc, 0.f);
    }
    for (int idx = tid; idx < 10 * HC; idx += 256) {      // zero pad rows 22..31
        int r = 22 + (idx >> 9);
        out1bf[r * HC + (idx & 511)] = (__bf16)0.f;
    }
    __syncthreads();

    // ---- GEMM2: (32x512) x (512x128) -> h2, 16 tiles x 16 K-steps ----
    for (int tt = wave; tt < 16; tt += 8) {
        int mt = tt >> 3, nt = tt & 7;
        v8f acc = vzero8();
        for (int kt = 0; kt < 16; ++kt) {
            if (kt + 1 < 16)
                __builtin_prefetch(W2bf + ((size_t)((kt + 1) * 8 + nt) * 32 + lane) * 16, 0, 0);
            v16bf a = load_a_frag(out1bf, mt * 16, kt * 32, HC, lane);
            v16bf b = load_b_frag(W2bf, kt * 8 + nt, lane);
            acc = wmma_bf16(a, b, acc);
        }
        int rbase = mt * 16 + ((lane & 16) ? 8 : 0);
        int col   = nt * 16 + (lane & 15);
#pragma unroll
        for (int v = 0; v < 8; ++v) h2s[(rbase + v) * CDIM + col] = acc[v];
    }
    __syncthreads();

    // ---- attention 2 (1 head) ----
    if (tid < A_NODES) {
        int j = tid; float s = 0.f, d = 0.f;
        for (int c = 0; c < CDIM; ++c) {
            float hv = h2s[j * CDIM + c];
            s += hv * att_src2[c]; d += hv * att_dst2[c];
        }
        asrc2[j] = s; adst2[j] = d;
    }
    __syncthreads();
    if (tid < A_NODES) {
        int i = tid;
        float di = adst2[i], ev[A_NODES], mx = -1e30f;
        for (int j = 0; j < A_NODES; ++j) {
            float z = di + asrc2[j];
            z = (z > 0.f) ? z : 0.2f * z;
            ev[j] = z; mx = fmaxf(mx, z);
        }
        float sum = 0.f;
        for (int j = 0; j < A_NODES; ++j) { float e = __expf(ev[j] - mx); ev[j] = e; sum += e; }
        float inv = 1.f / sum;
        for (int j = 0; j < A_NODES; ++j) alpha2[i * 22 + j] = ev[j] * inv;
    }
    __syncthreads();

    // ---- aggregate + bias + relu + mean-pool + sequence mask ----
    if (tid < CDIM) {
        int c = tid; float acc = 0.f;
        for (int i = 0; i < A_NODES; ++i) {
            float s = 0.f;
            for (int j = 0; j < A_NODES; ++j) s += alpha2[i * 22 + j] * h2s[j * CDIM + c];
            acc += fmaxf(s + b2[c], 0.f);
        }
        float pv = acc * (1.f / (float)A_NODES);
        int b = frame >> 8, t = frame & (TSTEPS - 1);
        if (t >= seqlen[b]) pv = 0.f;
        pooled[(size_t)frame * CDIM + c] = pv;
    }
}

// ---------------------------------------------------------------------------
// Persistent 2-layer masked LSTM + classifier.
// The recurrence is independent per batch row, so the 32 rows are split into
// two 16-row workgroups (one WMMA M-tile each) running on separate WGPs:
// halves WMMA-pipe pressure per SIMD on the latency-critical sequential scan.
// Wave w owns a fixed 16(row)x16(gate-col) patch across all 4 gates; the
// i/f/g/o D-fragments and cell state c live in registers (no gate buffer).
// ---------------------------------------------------------------------------
__global__ __launch_bounds__(512) void lstm_head(
    const float* __restrict__ pooled, const int* __restrict__ seqlen,
    const __bf16* __restrict__ Wc0, const __bf16* __restrict__ Wc1,
    const float* __restrict__ bih0, const float* __restrict__ bhh0,
    const float* __restrict__ bih1, const float* __restrict__ bhh1,
    const float* __restrict__ clf_w, const float* __restrict__ clf_b,
    float* __restrict__ out)
{
    __shared__ __bf16 hA0[16 * 384];   // [x_t(128) | h0(256)] bf16, layer-0 A
    __shared__ __bf16 hA1[16 * 512];   // [h0_t(256) | h1(256)] bf16, layer-1 A
    __shared__ float  hlast[16 * 256]; // masked layer-1 h (f32) for classifier
    __shared__ float  bs0[G4], bs1[G4];
    __shared__ int    sl[16];

    const int tid = threadIdx.x, lane = tid & 31, wave = tid >> 5;
    const int row0 = blockIdx.x * 16;              // this block's batch rows

    for (int i = tid; i < 16 * 384; i += 512) hA0[i] = (__bf16)0.f;
    for (int i = tid; i < 16 * 512; i += 512) hA1[i] = (__bf16)0.f;
    for (int i = tid; i < 16 * 256; i += 512) hlast[i] = 0.f;
    for (int i = tid; i < G4; i += 512) {
        bs0[i] = bih0[i] + bhh0[i];
        bs1[i] = bih1[i] + bhh1[i];
    }
    if (tid < 16) sl[tid] = seqlen[row0 + tid];
    __syncthreads();

    const int n16   = wave;                        // 0..15 (hidden col tile)
    const int col   = n16 * 16 + (lane & 15);      // hidden index 0..255
    const int rbase = (lane & 16) ? 8 : 0;

    float c0r[8], c1r[8];
    int   slr[8];
#pragma unroll
    for (int v = 0; v < 8; ++v) { c0r[v] = 0.f; c1r[v] = 0.f; slr[v] = sl[rbase + v]; }

    for (int t = 0; t < TSTEPS; ++t) {
        // stage x_t into hA0 cols [0,128)
        for (int i = tid; i < 16 * 128; i += 512) {
            int r = i >> 7, c = i & 127;
            hA0[r * 384 + c] = (__bf16)pooled[((size_t)(row0 + r) * TSTEPS + t) * CDIM + c];
        }
        __syncthreads();

        // layer-0 gates: K = 384 (12 k-tiles)
        v8f Di = vzero8(), Df = vzero8(), Dg = vzero8(), Do = vzero8();
        for (int kt = 0; kt < 12; ++kt) {
            if (kt + 1 < 12)
                __builtin_prefetch(Wc0 + ((size_t)((kt + 1) * 64 + n16) * 32 + lane) * 16, 0, 0);
            v16bf a = load_a_frag(hA0, 0, kt * 32, 384, lane);
            Di = wmma_bf16(a, load_b_frag(Wc0, kt * 64 +      n16, lane), Di);
            Df = wmma_bf16(a, load_b_frag(Wc0, kt * 64 + 16 + n16, lane), Df);
            Dg = wmma_bf16(a, load_b_frag(Wc0, kt * 64 + 32 + n16, lane), Dg);
            Do = wmma_bf16(a, load_b_frag(Wc0, kt * 64 + 48 + n16, lane), Do);
        }
        __syncthreads();   // all reads of hA0 done before h0 rewrite

#pragma unroll
        for (int v = 0; v < 8; ++v) {
            int row = rbase + v;
            float ig = sigf(Di[v] + bs0[col]);
            float fg = sigf(Df[v] + bs0[256 + col]);
            float gg = tanhf(Dg[v] + bs0[512 + col]);
            float og = sigf(Do[v] + bs0[768 + col]);
            float cn = fg * c0r[v] + ig * gg;
            float hn = og * tanhf(cn);
            if (t < slr[v]) {
                c0r[v] = cn;
                __bf16 hb = (__bf16)hn;
                hA0[row * 384 + 128 + col] = hb;   // recurrent input
                hA1[row * 512 + col]       = hb;   // layer-1 input
            }
        }
        __syncthreads();

        // layer-1 gates: K = 512 (16 k-tiles)
        Di = vzero8(); Df = vzero8(); Dg = vzero8(); Do = vzero8();
        for (int kt = 0; kt < 16; ++kt) {
            if (kt + 1 < 16)
                __builtin_prefetch(Wc1 + ((size_t)((kt + 1) * 64 + n16) * 32 + lane) * 16, 0, 0);
            v16bf a = load_a_frag(hA1, 0, kt * 32, 512, lane);
            Di = wmma_bf16(a, load_b_frag(Wc1, kt * 64 +      n16, lane), Di);
            Df = wmma_bf16(a, load_b_frag(Wc1, kt * 64 + 16 + n16, lane), Df);
            Dg = wmma_bf16(a, load_b_frag(Wc1, kt * 64 + 32 + n16, lane), Dg);
            Do = wmma_bf16(a, load_b_frag(Wc1, kt * 64 + 48 + n16, lane), Do);
        }
        __syncthreads();   // all reads of hA1 done before h1 rewrite

#pragma unroll
        for (int v = 0; v < 8; ++v) {
            int row = rbase + v;
            float ig = sigf(Di[v] + bs1[col]);
            float fg = sigf(Df[v] + bs1[256 + col]);
            float gg = tanhf(Dg[v] + bs1[512 + col]);
            float og = sigf(Do[v] + bs1[768 + col]);
            float cn = fg * c1r[v] + ig * gg;
            float hn = og * tanhf(cn);
            if (t < slr[v]) {
                c1r[v] = cn;
                hA1[row * 512 + 256 + col] = (__bf16)hn;
                hlast[row * 256 + col]     = hn;
            }
        }
        // next-iteration x-stage barrier orders these writes for the next step
    }
    __syncthreads();

    if (tid < 16) {
        float acc = clf_b[0];
        const float* hr = &hlast[tid * 256];
        for (int n = 0; n < HID; ++n) acc += hr[n] * clf_w[n];
        out[row0 + tid] = acc;
    }
}

// ---------------------------------------------------------------------------
// Launch
// ---------------------------------------------------------------------------
extern "C" void kernel_launch(void* const* d_in, const int* in_sizes, int n_in,
                              void* d_out, int out_size, void* d_ws, size_t ws_size,
                              hipStream_t stream) {
    (void)in_sizes; (void)n_in; (void)out_size; (void)ws_size;

    const float* feat     = (const float*)d_in[0];
    const int*   seqlen   = (const int*)  d_in[1];
    const float* W1       = (const float*)d_in[2];
    const float* att_src1 = (const float*)d_in[3];
    const float* att_dst1 = (const float*)d_in[4];
    const float* b1       = (const float*)d_in[5];
    const float* W2       = (const float*)d_in[6];
    const float* att_src2 = (const float*)d_in[7];
    const float* att_dst2 = (const float*)d_in[8];
    const float* b2       = (const float*)d_in[9];
    const float* Wih0     = (const float*)d_in[10];
    const float* Whh0     = (const float*)d_in[11];
    const float* bih0     = (const float*)d_in[12];
    const float* bhh0     = (const float*)d_in[13];
    const float* Wih1     = (const float*)d_in[14];
    const float* Whh1     = (const float*)d_in[15];
    const float* bih1     = (const float*)d_in[16];
    const float* bhh1     = (const float*)d_in[17];
    const float* clf_w    = (const float*)d_in[18];
    const float* clf_b    = (const float*)d_in[19];

    char* ws = (char*)d_ws;
    // packed bf16 weight fragments + pooled buffer (~6.2 MB total)
    __bf16* W1bf = (__bf16*)(ws + 0);            //  32 KB
    __bf16* W2bf = (__bf16*)(ws + 32768);        // 128 KB
    __bf16* Wc0  = (__bf16*)(ws + 163840);       // 768 KB (12x64 frags)
    __bf16* Wc1  = (__bf16*)(ws + 950272);       //   1 MB (16x64 frags)
    float*  pooled = (float*)(ws + 1998848);     //   4 MB [B,T,128]

    pack_w1<<<(32 * 32 * 16) / 256, 256, 0, stream>>>(W1, W1bf);
    pack_w2<<<(16 * 8 * 32 * 16) / 256, 256, 0, stream>>>(W2, W2bf);
    pack_cat<<<(12 * 64 * 32 * 16) / 256, 256, 0, stream>>>(Wih0, Whh0, 128, 384, Wc0);
    pack_cat<<<(16 * 64 * 32 * 16) / 256, 256, 0, stream>>>(Wih1, Whh1, 256, 512, Wc1);

    gat_frame<<<NFRAMES, 256, 0, stream>>>(feat, seqlen, W1bf, att_src1, att_dst1, b1,
                                           W2bf, att_src2, att_dst2, b2, pooled);

    lstm_head<<<2, 512, 0, stream>>>(pooled, seqlen, Wc0, Wc1,
                                     bih0, bhh0, bih1, bhh1,
                                     clf_w, clf_b, (float*)d_out);
}
